// TransformerBlock_86431921865283
// MI455X (gfx1250) — matrix operand order repaired and checked by reading, not compiled, not run on hardware
//
#include <hip/hip_runtime.h>
#include <hip/hip_bf16.h>

typedef __attribute__((ext_vector_type(16))) __bf16 v16bf;
typedef __attribute__((ext_vector_type(8)))  float  v8f;

// Problem constants (match reference)
#define kB 2
#define kS 2048
#define kD 1024
#define kH 16
#define kDH 64
#define kM (kB * kS)      // 4096 rows
#define kFF (4 * kD)      // 4096

// ---------------------------------------------------------------------------
// LDS-tiled weight convert + transpose: Wt[n*K + k] = bf16(W[k*N + n])
// grid = (N/32, K/32), block = (32, 8). Coalesced load and store.
// ---------------------------------------------------------------------------
__global__ __launch_bounds__(256)
void conv_transpose(const float* __restrict__ W, __bf16* __restrict__ Wt,
                    int K, int N) {
  __shared__ __bf16 tile[32][33];
  int n0 = blockIdx.x << 5;
  int k0 = blockIdx.y << 5;
#pragma unroll
  for (int r = threadIdx.y; r < 32; r += 8)
    tile[r][threadIdx.x] = (__bf16)W[(size_t)(k0 + r) * N + n0 + threadIdx.x];
  __syncthreads();
#pragma unroll
  for (int r = threadIdx.y; r < 32; r += 8)
    Wt[(size_t)(n0 + r) * K + k0 + threadIdx.x] = tile[threadIdx.x][r];
}

// ---------------------------------------------------------------------------
// LayerNorm (torch-style: unbiased std, eps added to std) -> fp32 and/or bf16
// One block (256 threads) per row of 1024.
// ---------------------------------------------------------------------------
__global__ __launch_bounds__(256)
void layernorm_k(const float* __restrict__ x, const float* __restrict__ scale,
                 const float* __restrict__ shift, float* __restrict__ outf,
                 __bf16* __restrict__ outb) {
  __shared__ float ssum[8], ssq[8];
  int row = blockIdx.x;
  const float* xr = x + (size_t)row * kD;
  float s = 0.f, s2 = 0.f;
  for (int i = threadIdx.x; i < kD; i += 256) {
    float v = xr[i];
    s += v; s2 += v * v;
  }
#pragma unroll
  for (int off = 1; off < 32; off <<= 1) {
    s  += __shfl_xor(s,  off, 32);
    s2 += __shfl_xor(s2, off, 32);
  }
  if ((threadIdx.x & 31) == 0) {
    ssum[threadIdx.x >> 5] = s;
    ssq [threadIdx.x >> 5] = s2;
  }
  __syncthreads();
  float ts = 0.f, ts2 = 0.f;
#pragma unroll
  for (int w = 0; w < 8; ++w) { ts += ssum[w]; ts2 += ssq[w]; }
  float mean  = ts * (1.0f / kD);
  float var   = (ts2 - (float)kD * mean * mean) * (1.0f / (kD - 1));
  float denom = sqrtf(fmaxf(var, 0.f)) + 1e-9f;
  float rden  = 1.0f / denom;
  for (int i = threadIdx.x; i < kD; i += 256) {
    float v = scale[i] * (xr[i] - mean) * rden + shift[i];
    size_t idx = (size_t)row * kD + i;
    if (outf) outf[idx] = v;
    if (outb) outb[idx] = (__bf16)v;
  }
}

// ---------------------------------------------------------------------------
// Fragment loads for the 32x64 wave tile (layouts per CDNA5 ISA tables).
// ---------------------------------------------------------------------------
__device__ __forceinline__ void ld_frags(const __bf16* __restrict__ Arow0,
                                         const __bf16* __restrict__ Arow1,
                                         const __bf16* __restrict__ Bt,
                                         int n0, int m, int hi, int K, int k0,
                                         v16bf& a0, v16bf& a1, v16bf* b) {
  const __bf16* pa0 = Arow0 + k0;
  const __bf16* pa1 = Arow1 + k0;
#pragma unroll
  for (int j = 0; j < 8; ++j) { a0[j] = pa0[j]; a0[8 + j] = pa0[16 + j]; }
#pragma unroll
  for (int j = 0; j < 8; ++j) { a1[j] = pa1[j]; a1[8 + j] = pa1[16 + j]; }
#pragma unroll
  for (int nf = 0; nf < 4; ++nf) {
    const __bf16* pb = Bt + (size_t)(n0 + (nf << 4) + m) * K + k0 + (hi << 3);
#pragma unroll
    for (int j = 0; j < 8; ++j) { b[nf][j] = pb[j]; b[nf][8 + j] = pb[16 + j]; }
  }
}

// ---------------------------------------------------------------------------
// bf16 WMMA GEMM: C[M,N] = A[M,K] @ Bt[N,K]^T (+bias)(+relu)(+resid)
// Each wave computes a 32x64 tile: 2 A-frags reused over 4 B-frags per K-step
// (8 WMMAs / 6 fragment loads). K-loop is unrolled x2 with ping-pong fragment
// buffers (no register rotation copies); loads for the next half-step overlap
// the current 8 WMMAs. Requires K % 64 == 0 (true for 1024/4096).
// Optional per-head-transposed V store: vtrans[((b*kH+h)*kDH + dh)*kS + s].
// ---------------------------------------------------------------------------
__global__ __launch_bounds__(256)
void gemm_bf16_wmma(const __bf16* __restrict__ A, const __bf16* __restrict__ Bt,
                    const float* __restrict__ bias,
                    const float* __restrict__ resid,
                    float* __restrict__ Cf, __bf16* __restrict__ Cb,
                    __bf16* __restrict__ vtrans,
                    int M, int N, int K, int relu) {
  int lane = threadIdx.x & 31;
  int wave = threadIdx.x >> 5;
  int tile = blockIdx.x * 8 + wave;
  int ntn = N >> 6;                      // 64-wide N tiles
  if (tile >= (M >> 5) * ntn) return;
  int m0 = (tile / ntn) << 5;            // 32-row M tiles
  int n0 = (tile % ntn) << 6;

  int m  = lane & 15;
  int hi = lane >> 4;

  v8f acc[2][4] = {};
  const __bf16* Arow0 = A + (size_t)(m0 + m) * K + (hi << 3);
  const __bf16* Arow1 = A + (size_t)(m0 + 16 + m) * K + (hi << 3);

  // ping-pong fragment buffers
  v16bf fa0[2], fa1[2], fb[2][4];
  ld_frags(Arow0, Arow1, Bt, n0, m, hi, K, 0, fa0[0], fa1[0], fb[0]);

  for (int k0 = 0; k0 < K; k0 += 64) {
    // ---- half-step A: compute buf0, load buf1 (k0+32 always valid) ----
    int k1 = k0 + 32;
    __builtin_prefetch(Arow0 + k1 + 32, 0, 3);
    __builtin_prefetch(Arow1 + k1 + 32, 0, 3);
    ld_frags(Arow0, Arow1, Bt, n0, m, hi, K, k1, fa0[1], fa1[1], fb[1]);
#pragma unroll
    for (int nf = 0; nf < 4; ++nf) {
      acc[0][nf] = __builtin_amdgcn_wmma_f32_16x16x32_bf16(
          false, fa0[0], false, fb[0][nf], (short)0, acc[0][nf], false, false);
      acc[1][nf] = __builtin_amdgcn_wmma_f32_16x16x32_bf16(
          false, fa1[0], false, fb[0][nf], (short)0, acc[1][nf], false, false);
    }
    // ---- half-step B: compute buf1, load buf0 (clamped on last iter) ----
    int k2 = (k0 + 64 < K) ? (k0 + 64) : k0;
    ld_frags(Arow0, Arow1, Bt, n0, m, hi, K, k2, fa0[0], fa1[0], fb[0]);
#pragma unroll
    for (int nf = 0; nf < 4; ++nf) {
      acc[0][nf] = __builtin_amdgcn_wmma_f32_16x16x32_bf16(
          false, fa0[1], false, fb[1][nf], (short)0, acc[0][nf], false, false);
      acc[1][nf] = __builtin_amdgcn_wmma_f32_16x16x32_bf16(
          false, fa1[1], false, fb[1][nf], (short)0, acc[1][nf], false, false);
    }
  }

#pragma unroll
  for (int half = 0; half < 2; ++half) {
    int mb = m0 + (half << 4) + (hi << 3);
#pragma unroll
    for (int nf = 0; nf < 4; ++nf) {
      int n = n0 + (nf << 4) + m;
      float bs = bias ? bias[n] : 0.f;
#pragma unroll
      for (int r = 0; r < 8; ++r) {
        int row = mb + r;
        size_t idx = (size_t)row * N + n;
        float v = acc[half][nf][r] + bs;
        if (relu) v = fmaxf(v, 0.f);
        if (resid) v += resid[idx];
        if (Cf) Cf[idx] = v;
        if (Cb) Cb[idx] = (__bf16)v;
        if (vtrans) {
          int bb = row >> 11;             // row / kS
          int ss = row & (kS - 1);
          int hh = n >> 6;                // n / kDH
          int nl = n & (kDH - 1);
          vtrans[(((size_t)bb * kH + hh) * kDH + nl) * kS + ss] = (__bf16)v;
        }
      }
    }
  }
}

// ---------------------------------------------------------------------------
// Causal flash attention, one wave per (b, h, 16-query tile).
// Q,K are bf16 [B*S][1024] heads packed; Vt is bf16 [B*H][64][S] (transposed).
// ---------------------------------------------------------------------------
__global__ __launch_bounds__(256)
void attn_fwd(const __bf16* __restrict__ Q, const __bf16* __restrict__ Km,
              const __bf16* __restrict__ Vt, __bf16* __restrict__ O) {
  __shared__ __bf16 plds[8][16][32];   // per-wave P staging (C-layout -> A-layout)
  int lane = threadIdx.x & 31;
  int wave = threadIdx.x >> 5;
  int wg = blockIdx.x * 8 + wave;
  if (wg >= kB * kH * (kS / 16)) return;
  int qt = wg & ((kS / 16) - 1);
  int bh = wg >> 7;                     // kS/16 == 128
  int b  = bh >> 4;
  int h  = bh & 15;
  int q0 = qt << 4;

  int m  = lane & 15;
  int hi = lane >> 4;

  const __bf16* qb = Q  + ((size_t)b * kS) * kD + h * kDH;
  const __bf16* kb = Km + ((size_t)b * kS) * kD + h * kDH;
  const __bf16* vt = Vt + (size_t)bh * kDH * kS;

  // Q A-frags for dh 0..31 and 32..63 (loaded once)
  v16bf aq0, aq1;
  {
    const __bf16* pq = qb + (size_t)(q0 + m) * kD + (hi << 4);
#pragma unroll
    for (int j = 0; j < 16; ++j) aq0[j] = pq[j];
#pragma unroll
    for (int j = 0; j < 16; ++j) aq1[j] = pq[32 + j];
  }

  float mrow[8], lrow[8], alpha[8];
#pragma unroll
  for (int r = 0; r < 8; ++r) { mrow[r] = -INFINITY; lrow[r] = 0.f; }
  v8f acc[4] = {};

  int qlast = q0 + 15;
  for (int kc = 0; kc <= qlast; kc += 32) {
    // scores for keys kc..kc+15 (s0) and kc+16..kc+31 (s1)
    v8f s0 = {}, s1 = {};
    {
      const __bf16* pk = kb + (size_t)(kc + m) * kD + (hi << 4);
      v16bf b0, b1;
#pragma unroll
      for (int j = 0; j < 16; ++j) { b0[j] = pk[j]; b1[j] = pk[32 + j]; }
      s0 = __builtin_amdgcn_wmma_f32_16x16x32_bf16(false, aq0, false, b0, (short)0, s0, false, false);
      s0 = __builtin_amdgcn_wmma_f32_16x16x32_bf16(false, aq1, false, b1, (short)0, s0, false, false);
    }
    {
      const __bf16* pk = kb + (size_t)(kc + 16 + m) * kD + (hi << 4);
      v16bf b0, b1;
#pragma unroll
      for (int j = 0; j < 16; ++j) { b0[j] = pk[j]; b1[j] = pk[32 + j]; }
      s1 = __builtin_amdgcn_wmma_f32_16x16x32_bf16(false, aq0, false, b0, (short)0, s1, false, false);
      s1 = __builtin_amdgcn_wmma_f32_16x16x32_bf16(false, aq1, false, b1, (short)0, s1, false, false);
    }

    // scale, causal mask, online softmax stats (half-wave row reductions)
#pragma unroll
    for (int r = 0; r < 8; ++r) {
      int qi = q0 + r + (hi << 3);
      float x0 = s0[r] * 0.125f; if (kc + m      > qi) x0 = -INFINITY;
      float x1 = s1[r] * 0.125f; if (kc + 16 + m > qi) x1 = -INFINITY;
      float mm = fmaxf(x0, x1);
#pragma unroll
      for (int off = 1; off < 16; off <<= 1) mm = fmaxf(mm, __shfl_xor(mm, off, 32));
      float mn = fmaxf(mrow[r], mm);
      float al = expf(mrow[r] - mn);
      float e0 = expf(x0 - mn);
      float e1 = expf(x1 - mn);
      float rs = e0 + e1;
#pragma unroll
      for (int off = 1; off < 16; off <<= 1) rs += __shfl_xor(rs, off, 32);
      lrow[r] = lrow[r] * al + rs;
      mrow[r] = mn;
      alpha[r] = al;
      plds[wave][r + (hi << 3)][m]      = (__bf16)e0;
      plds[wave][r + (hi << 3)][16 + m] = (__bf16)e1;
    }

    // cross-lane LDS handoff: ensure stores land before swizzled reads
    asm volatile("s_wait_dscnt 0x0" ::: "memory");

    // P A-frag (16x32) from LDS
    v16bf ap;
    {
      const __bf16* pl = &plds[wave][m][hi << 4];
#pragma unroll
      for (int j = 0; j < 16; ++j) ap[j] = pl[j];
    }

    // P @ V : 4 B-frags over dh (contiguous along keys in Vt), rescale accs
#pragma unroll
    for (int nf = 0; nf < 4; ++nf) {
#pragma unroll
      for (int r = 0; r < 8; ++r) acc[nf][r] *= alpha[r];
      const __bf16* pv = vt + (size_t)((nf << 4) + m) * kS + kc + (hi << 4);
      v16bf bvf;
#pragma unroll
      for (int j = 0; j < 16; ++j) bvf[j] = pv[j];
      acc[nf] = __builtin_amdgcn_wmma_f32_16x16x32_bf16(
          false, ap, false, bvf, (short)0, acc[nf], false, false);
    }
  }

  // normalize and write merged-head bf16 output
#pragma unroll
  for (int nf = 0; nf < 4; ++nf) {
#pragma unroll
    for (int r = 0; r < 8; ++r) {
      float o = acc[nf][r] / lrow[r];
      size_t idx = ((size_t)b * kS + q0 + r + (hi << 3)) * kD + h * kDH + (nf << 4) + m;
      O[idx] = (__bf16)o;
    }
  }
}

// ---------------------------------------------------------------------------
// Launch orchestration
// ---------------------------------------------------------------------------
extern "C" void kernel_launch(void* const* d_in, const int* in_sizes, int n_in,
                              void* d_out, int out_size, void* d_ws, size_t ws_size,
                              hipStream_t stream) {
  const float* inputs    = (const float*)d_in[0];
  const float* ln1_scale = (const float*)d_in[1];
  const float* ln1_shift = (const float*)d_in[2];
  const float* Wq = (const float*)d_in[3];  const float* bq = (const float*)d_in[4];
  const float* Wk = (const float*)d_in[5];  const float* bk = (const float*)d_in[6];
  const float* Wv = (const float*)d_in[7];  const float* bv = (const float*)d_in[8];
  const float* Wo = (const float*)d_in[9];  const float* bo = (const float*)d_in[10];
  const float* W1 = (const float*)d_in[11]; const float* b1 = (const float*)d_in[12];
  const float* W2 = (const float*)d_in[13]; const float* b2 = (const float*)d_in[14];
  const float* ln2_scale = (const float*)d_in[15];
  const float* ln2_shift = (const float*)d_in[16];

  // workspace carve-up
  char* ws = (char*)d_ws;
  size_t off = 0;
  auto alloc = [&](size_t bytes) { char* p = ws + off; off += (bytes + 255) & ~(size_t)255; return p; };

  __bf16* xb    = (__bf16*)alloc((size_t)kM * kD * 2);     // ln1 out, bf16
  __bf16* qbuf  = (__bf16*)alloc((size_t)kM * kD * 2);
  __bf16* kbuf  = (__bf16*)alloc((size_t)kM * kD * 2);
  __bf16* vtb   = (__bf16*)alloc((size_t)kM * kD * 2);     // V, [B*H][64][S]
  __bf16* abuf  = (__bf16*)alloc((size_t)kM * kD * 2);     // attention out, bf16
  float*  o1f   = (float*) alloc((size_t)kM * kD * 4);     // attn proj + residual
  float*  o2f   = (float*) alloc((size_t)kM * kD * 4);     // ln2 out fp32
  __bf16* o2b   = (__bf16*)alloc((size_t)kM * kD * 2);     // ln2 out bf16
  __bf16* hidb  = (__bf16*)alloc((size_t)kM * kFF * 2);    // ffn hidden bf16
  __bf16* WqT   = (__bf16*)alloc((size_t)kD * kD * 2);
  __bf16* WkT   = (__bf16*)alloc((size_t)kD * kD * 2);
  __bf16* WvT   = (__bf16*)alloc((size_t)kD * kD * 2);
  __bf16* WoT   = (__bf16*)alloc((size_t)kD * kD * 2);
  __bf16* W1T   = (__bf16*)alloc((size_t)kD * kFF * 2);
  __bf16* W2T   = (__bf16*)alloc((size_t)kFF * kD * 2);

  // 1) weight convert + transpose (LDS-tiled; Wt[N][K] from W[K][N])
  {
    dim3 blk(32, 8);
    conv_transpose<<<dim3(kD  / 32, kD  / 32), blk, 0, stream>>>(Wq, WqT, kD, kD);
    conv_transpose<<<dim3(kD  / 32, kD  / 32), blk, 0, stream>>>(Wk, WkT, kD, kD);
    conv_transpose<<<dim3(kD  / 32, kD  / 32), blk, 0, stream>>>(Wv, WvT, kD, kD);
    conv_transpose<<<dim3(kD  / 32, kD  / 32), blk, 0, stream>>>(Wo, WoT, kD, kD);
    conv_transpose<<<dim3(kFF / 32, kD  / 32), blk, 0, stream>>>(W1, W1T, kD, kFF);
    conv_transpose<<<dim3(kD  / 32, kFF / 32), blk, 0, stream>>>(W2, W2T, kFF, kD);
  }

  // 2) LN1 -> bf16
  layernorm_k<<<kM, 256, 0, stream>>>(inputs, ln1_scale, ln1_shift, nullptr, xb);

  // 3) QKV projections (M=4096, N=1024, K=1024): 2048 wave-tiles -> 256 blocks
  int g_sq = ((kM >> 5) * (kD >> 6)) / 8;    // 256
  gemm_bf16_wmma<<<g_sq, 256, 0, stream>>>(xb, WqT, bq, nullptr, nullptr, qbuf, nullptr, kM, kD, kD, 0);
  gemm_bf16_wmma<<<g_sq, 256, 0, stream>>>(xb, WkT, bk, nullptr, nullptr, kbuf, nullptr, kM, kD, kD, 0);
  gemm_bf16_wmma<<<g_sq, 256, 0, stream>>>(xb, WvT, bv, nullptr, nullptr, nullptr, vtb, kM, kD, kD, 0);

  // 4) causal attention: B*H*(S/16) = 4096 waves -> 512 blocks
  attn_fwd<<<(kB * kH * (kS / 16)) / 8, 256, 0, stream>>>(qbuf, kbuf, vtb, abuf);

  // 5) output projection + input residual -> o1 (fp32)
  gemm_bf16_wmma<<<g_sq, 256, 0, stream>>>(abuf, WoT, bo, inputs, o1f, nullptr, nullptr, kM, kD, kD, 0);

  // 6) LN2 -> fp32 + bf16
  layernorm_k<<<kM, 256, 0, stream>>>(o1f, ln2_scale, ln2_shift, o2f, o2b);

  // 7) FFN1 with ReLU (M=4096, N=4096, K=1024) -> bf16 hidden
  int g_ff1 = ((kM >> 5) * (kFF >> 6)) / 8;  // 1024
  gemm_bf16_wmma<<<g_ff1, 256, 0, stream>>>(o2b, W1T, b1, nullptr, nullptr, hidb, nullptr, kM, kFF, kD, 1);

  // 8) FFN2 + o2 residual (M=4096, N=1024, K=4096) -> d_out fp32
  gemm_bf16_wmma<<<g_sq, 256, 0, stream>>>(hidb, W2T, b2, o2f, (float*)d_out, nullptr, nullptr, kM, kD, kFF, 0);

  (void)in_sizes; (void)n_in; (void)out_size; (void)ws_size;
}
